// LSTM_14345190768817
// MI455X (gfx1250) — matrix operand order repaired and checked by reading, not compiled, not run on hardware
//
#include <hip/hip_runtime.h>
#include <hip/hip_bf16.h>

// ---------------------------------------------------------------------------
// LSTM scan for MI455X (gfx1250, wave32).
// B=256, T=1024, D=128, H=256.  Per step: 4 GEMMs [256,384]x[384,256] (bf16
// WMMA, f32 accum) + gate elementwise.  Persistent kernel, one grid barrier
// per timestep, cell state held in registers for the whole scan.
// Round 2: dual WMMA accumulator chains (ILP) + async global->LDS staging of
// the recurrent h tile (ASYNCcnt path, gfx1250-specific).
// ---------------------------------------------------------------------------

typedef __attribute__((ext_vector_type(16))) __bf16 v16bf;
typedef __attribute__((ext_vector_type(8)))  __bf16 v8bf;
typedef __attribute__((ext_vector_type(8)))  float  v8f;

#define B_    256
#define T_    1024
#define D_    128
#define H_    256
#define KD_   384      // D + H
#define KP_   392      // padded LDS row stride in bf16 elems (784B -> 4-DWORD bank skew)
#define NWG   256      // grid: 16 b-tiles x 16 h-tiles
#define NTHR  128      // 4 waves: one gate tile per wave

__device__ __forceinline__ float sigmoid_f(float x) {
  return 1.0f / (1.0f + __expf(-x));
}
__device__ __forceinline__ float tanh_f(float x) {
  float e = __expf(2.0f * x);
  return 1.0f - 2.0f / (e + 1.0f);   // safe at +/-inf
}

// Low 32 bits of a generic pointer into __shared__ == wave-relative LDS byte
// offset (ISA 10.2: LDS aperture; the low 32 address bits give the LDS address).
__device__ __forceinline__ unsigned lds_off32(const void* p) {
  return (unsigned)(size_t)p;
}

__global__ __launch_bounds__(NTHR)
void lstm_persistent(const float* __restrict__ x,
                     const float* __restrict__ Wi, const float* __restrict__ bi,
                     const float* __restrict__ Wf, const float* __restrict__ bff,
                     const float* __restrict__ Wo, const float* __restrict__ bo,
                     const float* __restrict__ Wc, const float* __restrict__ bc,
                     const float* __restrict__ W2,
                     float*       __restrict__ dout,      // [256 out][65536 hid]
                     unsigned*    __restrict__ counter,   // zeroed before launch
                     float*       __restrict__ partial,   // [B_][16] head partials
                     __bf16*      __restrict__ hbuf0,     // [B_][H_] bf16
                     __bf16*      __restrict__ hbuf1)     // [B_][H_] bf16
{
  __shared__ __bf16 wlds[4][16][KP_];   // per-gate weight slice, bf16, row-major [h][k]
  __shared__ __bf16 clds[16][KP_];      // comb tile [b][k]
  __shared__ float  dlds[4][256];       // gate pre-activations, row-major 16x16
  __shared__ float  blds[4][16];        // bias slices

  const int tid  = threadIdx.x;
  const int wave = tid >> 5;            // 0..3 -> gate
  const int lane = tid & 31;
  const int wg   = blockIdx.x;
  const int b0   = (wg & 15) * 16;      // b-tile origin
  const int h0   = (wg >> 4) * 16;      // h-tile origin

  const float* Wg[4] = {Wi, Wf, Wo, Wc};
  const float* bg[4] = {bi, bff, bo, bc};

  // ---- one-time: weights + biases -> LDS (f32 -> bf16) --------------------
  for (int g = 0; g < 4; ++g) {
    const float* Wsrc = Wg[g] + (size_t)h0 * KD_;
    for (int idx = tid; idx < 16 * KD_; idx += NTHR) {
      int r = idx / KD_, k = idx - r * KD_;
      wlds[g][r][k] = (__bf16)Wsrc[r * KD_ + k];
    }
    if (tid < 16) blds[g][tid] = bg[g][h0 + tid];
  }

  // ---- zero my slice of h_{-1} (buffer 0) ---------------------------------
  for (int e = tid; e < 256; e += NTHR) {
    int m = e >> 4, n = e & 15;
    hbuf0[(size_t)(b0 + m) * H_ + (h0 + n)] = (__bf16)0.0f;
  }

  float creg0 = 0.0f, creg1 = 0.0f;     // cell state: 2 elems per thread
  unsigned barCnt = 0;

  auto grid_barrier = [&]() {
    __threadfence();
    __syncthreads();
    if (tid == 0) {
      ++barCnt;
      __hip_atomic_fetch_add(counter, 1u, __ATOMIC_RELEASE, __HIP_MEMORY_SCOPE_AGENT);
      const unsigned target = barCnt * NWG;
      while (__hip_atomic_load(counter, __ATOMIC_ACQUIRE, __HIP_MEMORY_SCOPE_AGENT) < target)
        __builtin_amdgcn_s_sleep(1);
    }
    __syncthreads();
    __threadfence();
  };

  grid_barrier();   // h_{-1} zeros visible everywhere

  const int gl = lane >> 4;             // WMMA lane-group (0: lanes 0-15, 1: 16-31)
  const int ml = lane & 15;             // row (A) / column (B,D) index

  for (int t = 0; t < T_; ++t) {
    const __bf16* hsrc = (t & 1) ? hbuf1 : hbuf0;
    __bf16*       hdst = (t & 1) ? hbuf0 : hbuf1;

    // ---- stage comb = [x_t | h_{t-1}] tile into LDS -----------------------
    {
      const int r      = tid >> 3;             // 16 rows, 8 threads per row

      // h part first: kick off async global->LDS b128 copies (no VGPR bounce,
      // overlaps with the f32->bf16 x conversion below).  ASYNCcnt-tracked.
      const int hstart = (tid & 7) * 32;       // 32 bf16 of h per thread
      const __bf16* hp = hsrc + (size_t)(b0 + r) * H_ + hstart;
      #pragma unroll
      for (int u = 0; u < 32; u += 8) {
        unsigned dst = lds_off32(&clds[r][D_ + hstart + u]);
        unsigned long long ga = (unsigned long long)(const void*)(hp + u);
        asm volatile("global_load_async_to_lds_b128 %0, %1, off"
                     :: "v"(dst), "v"(ga) : "memory");
      }

      // x part: load f32, convert, pack to bf16x8, store to LDS.
      const int dstart = (tid & 7) * 16;       // 16 f32 of x per thread
      const float* xp  = x + ((size_t)(b0 + r) * T_ + t) * D_ + dstart;
      #pragma unroll
      for (int u = 0; u < 16; u += 8) {
        float4 va = *(const float4*)(xp + u);
        float4 vb = *(const float4*)(xp + u + 4);
        v8bf pk;
        pk[0] = (__bf16)va.x; pk[1] = (__bf16)va.y;
        pk[2] = (__bf16)va.z; pk[3] = (__bf16)va.w;
        pk[4] = (__bf16)vb.x; pk[5] = (__bf16)vb.y;
        pk[6] = (__bf16)vb.z; pk[7] = (__bf16)vb.w;
        *(v8bf*)&clds[r][dstart + u] = pk;
      }

      // async LDS writes are NOT covered by DScnt/barrier: drain ASYNCcnt.
      asm volatile("s_wait_asynccnt 0x0" ::: "memory");
    }
    __syncthreads();

    // ---- GEMM: wave `wave` computes gate tile, K=384 = 12 x k32 -----------
    // Two independent accumulator chains so consecutive WMMAs overlap in the
    // matrix pipe instead of serializing on the C operand.
    v8f acc0 = {}, acc1 = {};
    #pragma unroll
    for (int kk = 0; kk < 12; ++kk) {
      const int k0 = kk * 32;
      v16bf afrag, bfrag;
      {
        // A 16x32 bf16: lane-group gl -> K offsets {gl*8..+7, 16+gl*8..+7}
        const __bf16* ap = &clds[ml][k0 + gl * 8];
        v8bf lo = *(const v8bf*)ap;
        v8bf hi = *(const v8bf*)(ap + 16);
        #pragma unroll
        for (int q = 0; q < 8; ++q) { afrag[q] = lo[q]; afrag[8 + q] = hi[q]; }
      }
      {
        // B 32x16 bf16: column n = ml, lane-group gl -> K offsets {gl*8..+7, 16+gl*8..+7}
        const __bf16* bp = &wlds[wave][ml][k0 + gl * 8];
        v8bf lo = *(const v8bf*)bp;
        v8bf hi = *(const v8bf*)(bp + 16);
        #pragma unroll
        for (int q = 0; q < 8; ++q) { bfrag[q] = lo[q]; bfrag[8 + q] = hi[q]; }
      }
      if (kk & 1)
        acc1 = __builtin_amdgcn_wmma_f32_16x16x32_bf16(
            false, afrag, false, bfrag, (short)0, acc1, false, false);
      else
        acc0 = __builtin_amdgcn_wmma_f32_16x16x32_bf16(
            false, afrag, false, bfrag, (short)0, acc0, false, false);
    }
    // D layout: lane gl/ml -> (m = r + 8*gl, n = ml)
    #pragma unroll
    for (int rr = 0; rr < 8; ++rr)
      dlds[wave][(rr + 8 * gl) * 16 + ml] = acc0[rr] + acc1[rr];
    __syncthreads();

    // ---- elementwise gates; update register cell state --------------------
    const bool last = (t == T_ - 1);
    float hv0, hv1;
    {
      const int e = tid * 2;
      const int n = e & 15;
      float ig = sigmoid_f(dlds[0][e] + blds[0][n]);
      float fg = sigmoid_f(dlds[1][e] + blds[1][n]);
      float og = sigmoid_f(dlds[2][e] + blds[2][n]);
      float ct = tanh_f   (dlds[3][e] + blds[3][n]);
      creg0 = fg * creg0 + ig * ct;
      hv0   = og * tanh_f(creg0);
      hdst[(size_t)(b0 + (e >> 4)) * H_ + (h0 + n)] = (__bf16)hv0;
    }
    {
      const int e = tid * 2 + 1;
      const int n = e & 15;
      float ig = sigmoid_f(dlds[0][e] + blds[0][n]);
      float fg = sigmoid_f(dlds[1][e] + blds[1][n]);
      float og = sigmoid_f(dlds[2][e] + blds[2][n]);
      float ct = tanh_f   (dlds[3][e] + blds[3][n]);
      creg1 = fg * creg1 + ig * ct;
      hv1   = og * tanh_f(creg1);
      hdst[(size_t)(b0 + (e >> 4)) * H_ + (h0 + n)] = (__bf16)hv1;
    }

    if (last) {
      const int e = tid * 2;
      const int m = e >> 4, n = e & 15;
      // final hid (f32) straight to d_out
      dout[256 + (size_t)(b0 + m) * H_ + (h0 + n)]     = hv0;
      dout[256 + (size_t)(b0 + m) * H_ + (h0 + n + 1)] = hv1;
      // head partial: sum over this WG's 16 h-columns for row m
      float part = hv0 * W2[h0 + n]     + creg0 * W2[H_ + h0 + n]
                 + hv1 * W2[h0 + n + 1] + creg1 * W2[H_ + h0 + n + 1];
      #pragma unroll
      for (int off = 4; off > 0; off >>= 1)
        part += __shfl_xor(part, off, 8);          // 8 lanes cover one row
      if ((tid & 7) == 0)
        partial[(size_t)(b0 + m) * 16 + (wg >> 4)] = part;  // unique slot: deterministic
    }

    grid_barrier();
  }
}

__global__ __launch_bounds__(256)
void lstm_head(const float* __restrict__ partial,
               const float* __restrict__ b2,
               float*       __restrict__ dout)
{
  const int b = threadIdx.x;
  float s = 0.0f;
  #pragma unroll
  for (int j = 0; j < 16; ++j) s += partial[(size_t)b * 16 + j];
  dout[b] = sigmoid_f(s + b2[0]);
}

extern "C" void kernel_launch(void* const* d_in, const int* in_sizes, int n_in,
                              void* d_out, int out_size, void* d_ws, size_t ws_size,
                              hipStream_t stream) {
  (void)in_sizes; (void)n_in; (void)out_size; (void)ws_size;

  const float* x  = (const float*)d_in[0];
  const float* Wi = (const float*)d_in[1];
  const float* bi = (const float*)d_in[2];
  const float* Wf = (const float*)d_in[3];
  const float* bf = (const float*)d_in[4];
  const float* Wo = (const float*)d_in[5];
  const float* bo = (const float*)d_in[6];
  const float* Wc = (const float*)d_in[7];
  const float* bc = (const float*)d_in[8];
  const float* W2 = (const float*)d_in[9];
  const float* b2 = (const float*)d_in[10];

  char* ws = (char*)d_ws;
  unsigned* counter = (unsigned*)ws;                       // @0, 256B reserved
  float*    partial = (float*)(ws + 256);                  // [256][16] f32 = 16KB
  __bf16*   hbuf0   = (__bf16*)(ws + 16896);               // [256][256] bf16 = 128KB
  __bf16*   hbuf1   = (__bf16*)(ws + 16896 + B_ * H_ * 2); // second buffer

  // zero barrier counter (graph-capturable); partial/hbuf are fully overwritten
  hipMemsetAsync(d_ws, 0, 256, stream);

  lstm_persistent<<<dim3(NWG), dim3(NTHR), 0, stream>>>(
      x, Wi, bi, Wf, bf, Wo, bo, Wc, bc, W2,
      (float*)d_out, counter, partial, hbuf0, hbuf1);

  lstm_head<<<dim3(1), dim3(256), 0, stream>>>(partial, b2, (float*)d_out);
}
